// S2T_D_64098091925732
// MI455X (gfx1250) — matrix operand order repaired and checked by reading, not compiled, not run on hardware
//
#include <hip/hip_runtime.h>
#include <hip/hip_bf16.h>
#include <math.h>

// Problem constants (from reference): SEQ=128, B=512, V=50257, D=300, H=256
#define SEQB   512
#define HID    256
#define GATES  1024          // 4*H
#define DIM    300
#define KX     320           // D padded to multiple of 32
#define KTOT   576           // KX + H  (18 chunks of 32)
#define NCHUNK 18

typedef __attribute__((ext_vector_type(16))) _Float16 v16h;
typedef __attribute__((ext_vector_type(8)))  float    v8f;
typedef int v4i __attribute__((vector_size(16)));

union Frag16 {
    v16h v;
    _Float16 e[16];
    unsigned int u[8];
};

#if defined(__has_builtin)
#if __has_builtin(__builtin_amdgcn_global_load_async_to_lds_b128)
#define HAVE_ASYNC_LDS 1
#endif
#endif
#ifndef HAVE_ASYNC_LDS
#define HAVE_ASYNC_LDS 0
#endif

#define AS1 __attribute__((address_space(1)))
#define AS3 __attribute__((address_space(3)))

// ---------------------------------------------------------------------------
// Prep: pack [W_ih | W_hh] into f16 Wcat (GATES x KTOT), columns 300..319 zero
// ---------------------------------------------------------------------------
__global__ void prep_wcat_kernel(const float* __restrict__ Wih,
                                 const float* __restrict__ Whh,
                                 _Float16* __restrict__ Wcat) {
    int idx = blockIdx.x * blockDim.x + threadIdx.x;
    if (idx >= GATES * KTOT) return;
    int n = idx / KTOT;
    int k = idx - n * KTOT;
    float v;
    if (k < KX) v = (k < DIM) ? Wih[n * DIM + k] : 0.0f;
    else        v = Whh[n * HID + (k - KX)];
    Wcat[idx] = (_Float16)v;
}

// Prep: bias sum + zero state (h double buffers f16, c f32)
__global__ void prep_state_kernel(const float* __restrict__ bih,
                                  const float* __restrict__ bhh,
                                  float* __restrict__ biasSum,
                                  _Float16* __restrict__ h0,
                                  _Float16* __restrict__ h1,
                                  float* __restrict__ cbuf) {
    int idx = blockIdx.x * blockDim.x + threadIdx.x;
    if (idx < GATES) biasSum[idx] = bih[idx] + bhh[idx];
    if (idx < SEQB * HID) {
        h0[idx] = (_Float16)0.0f;
        h1[idx] = (_Float16)0.0f;
        cbuf[idx] = 0.0f;
    }
}

// ---------------------------------------------------------------------------
// One LSTM timestep, fused:
//   gates = [x_t | h_prev] @ Wcat^T + bias   (WMMA f16 -> f32, K = 576)
//   i,f,g,o -> c,h update (pointwise, in-register)
// Grid: 64 blocks x 256 threads = 512 waves.
//   wave id w: tileM = w & 31 (batch tile), ngrp = w >> 5 (hidden tile)
//   => all 8 waves of a block share ngrp, so the block's weight slice
//      (4 gates x 16 rows x 576 cols f16 = 72 KB) is staged in LDS once.
// ---------------------------------------------------------------------------
__global__ void __launch_bounds__(256)
lstm_step_kernel(const int*      __restrict__ tokens,   // B tokens for this t
                 const float*    __restrict__ emb,      // (V, 300) f32
                 const _Float16* __restrict__ Wcat,     // (1024, 576) f16
                 const float*    __restrict__ biasSum,  // (1024)
                 const _Float16* __restrict__ hPrev,    // (512, 256) f16
                 _Float16*       __restrict__ hNext,    // (512, 256) f16
                 float*          __restrict__ cBuf) {   // (512, 256) f32
    // LDS weight stage: [kc][g*16+row][32 cols] f16 = 73728 B
    __shared__ __align__(16) _Float16 sW[NCHUNK * 64 * 32];

    const int lane  = threadIdx.x & 31;
    const int l15   = lane & 15;
    const int half  = lane >> 4;
    const int ngrp  = blockIdx.x >> 2;                       // shared by block
    const int tileM = (blockIdx.x * 8 + (threadIdx.x >> 5)) & 31;

    // ---- Stage the block's Wcat slice into LDS (4608 x 16B chunks) ----
    for (int i = 0; i < NCHUNK; ++i) {
        const int c    = threadIdx.x + i * 256;
        const int kc   = c >> 8;            // 256 chunks per k-chunk
        const int rem  = c & 255;
        const int grow = rem >> 2;          // g*16 + row
        const int part = rem & 3;           // 8 f16 per part
        const int g    = grow >> 4;
        const int r    = grow & 15;
        const _Float16* src = Wcat + (g * HID + ngrp * 16 + r) * KTOT
                                   + kc * 32 + part * 8;
        _Float16* dst = sW + (kc * 64 + grow) * 32 + part * 8;
#if HAVE_ASYNC_LDS
        __builtin_amdgcn_global_load_async_to_lds_b128(
            (AS1 v4i*)src, (AS3 v4i*)dst, 0, 0);
#else
        *(uint4*)dst = *(const uint4*)src;
#endif
    }
#if HAVE_ASYNC_LDS
    asm volatile("s_wait_asynccnt 0" ::: "memory");
#endif
    __syncthreads();

    // A-fragment row (batch index) for this lane (16-bit A layout: M = lane&15)
    const int mA = tileM * 16 + l15;
    const int tok = tokens[mA];
    const float*    __restrict__ xrow = emb + (long)tok * DIM;
    const _Float16* __restrict__ hrow = hPrev + mA * HID;

    // Accumulators initialized with bias (C layout: n = lane&15, both halves)
    v8f acc[4];
#pragma unroll
    for (int g = 0; g < 4; ++g) {
        const float bv = biasSum[g * HID + ngrp * 16 + l15];
#pragma unroll
        for (int r = 0; r < 8; ++r) acc[g][r] = bv;
    }

#pragma unroll
    for (int kc = 0; kc < NCHUNK; ++kc) {
        const int kbase = kc * 32;

        // --- A fragment ---
        // slot v: K pair (v<4 -> 2v, v>=4 -> 16+2(v-4)) + 8*half
        // => lane covers two contiguous 16B blocks:
        //    [kbase+8h, +7] -> e[0..7],  [kbase+16+8h, +7] -> e[8..15]
        Frag16 a;
        if (kbase + 32 <= DIM) {
            // pure x region (kc 0..8), 16B-aligned float4 loads + pack
            const float4 x0 = *(const float4*)(xrow + kbase + 8 * half);
            const float4 x1 = *(const float4*)(xrow + kbase + 8 * half + 4);
            const float4 x2 = *(const float4*)(xrow + kbase + 16 + 8 * half);
            const float4 x3 = *(const float4*)(xrow + kbase + 16 + 8 * half + 4);
            a.e[0]  = (_Float16)x0.x; a.e[1]  = (_Float16)x0.y;
            a.e[2]  = (_Float16)x0.z; a.e[3]  = (_Float16)x0.w;
            a.e[4]  = (_Float16)x1.x; a.e[5]  = (_Float16)x1.y;
            a.e[6]  = (_Float16)x1.z; a.e[7]  = (_Float16)x1.w;
            a.e[8]  = (_Float16)x2.x; a.e[9]  = (_Float16)x2.y;
            a.e[10] = (_Float16)x2.z; a.e[11] = (_Float16)x2.w;
            a.e[12] = (_Float16)x3.x; a.e[13] = (_Float16)x3.y;
            a.e[14] = (_Float16)x3.z; a.e[15] = (_Float16)x3.w;
        } else if (kbase < KX) {
            // boundary chunk kc=9 (K 288..319): second block always >= 304 -> 0
#pragma unroll
            for (int i = 0; i < 8; ++i) {
                const int k = kbase + 8 * half + i;
                a.e[i]     = (k < DIM) ? (_Float16)xrow[k] : (_Float16)0.0f;
                a.e[8 + i] = (_Float16)0.0f;
            }
        } else {
            // h region: direct f16, two 16B loads
            const _Float16* p = hrow + (kbase - KX) + 8 * half;
            *(uint4*)&a.u[0] = *(const uint4*)p;
            *(uint4*)&a.u[4] = *(const uint4*)(p + 16);
        }

        // --- B fragments from LDS (2 x ds_load_b128 each) + WMMA per gate ---
#pragma unroll
        for (int g = 0; g < 4; ++g) {
            const _Float16* wsrc = sW + (kc * 64 + g * 16 + l15) * 32 + half * 8;
            Frag16 b;
            *(uint4*)&b.u[0] = *(const uint4*)wsrc;
            *(uint4*)&b.u[4] = *(const uint4*)(wsrc + 16);
            acc[g] = __builtin_amdgcn_wmma_f32_16x16x32_f16(
                false, a.v, false, b.v, (short)0, acc[g], false, false);
        }
    }

    // --- Pointwise gate math + state update ---
    // C/D layout: lanes 0-15: M = r, N = lane; lanes 16-31: M = 8+r, N = lane-16
    const int mC = tileM * 16 + half * 8;
    const int nC = ngrp * 16 + l15;
#pragma unroll
    for (int r = 0; r < 8; ++r) {
        const int idx = (mC + r) * HID + nC;
        const float iv = 1.0f / (1.0f + __expf(-acc[0][r]));
        const float fv = 1.0f / (1.0f + __expf(-acc[1][r]));
        const float gv = tanhf(acc[2][r]);
        const float ov = 1.0f / (1.0f + __expf(-acc[3][r]));
        const float cv = fv * cBuf[idx] + iv * gv;
        cBuf[idx] = cv;
        hNext[idx] = (_Float16)(ov * tanhf(cv));
    }
}

// ---------------------------------------------------------------------------
// Tail MLP: score[b] = relu(h @ W1^T + b1) @ W2^T + b2   (33 MFLOP, scalar)
// ---------------------------------------------------------------------------
__global__ void mlp_kernel(const _Float16* __restrict__ hFin,
                           const float* __restrict__ W1,
                           const float* __restrict__ b1,
                           const float* __restrict__ W2,
                           const float* __restrict__ b2,
                           float* __restrict__ out) {
    int b = blockIdx.x * blockDim.x + threadIdx.x;
    if (b >= SEQB) return;
    const _Float16* __restrict__ h = hFin + b * HID;
    float acc = b2[0];
    for (int j = 0; j < 128; ++j) {
        const float* __restrict__ w = W1 + j * HID;
        float s = b1[j];
#pragma unroll 8
        for (int k = 0; k < HID; ++k) s += (float)h[k] * w[k];
        acc += fmaxf(s, 0.0f) * W2[j];
    }
    out[b] = acc;
}

// ---------------------------------------------------------------------------
extern "C" void kernel_launch(void* const* d_in, const int* in_sizes, int n_in,
                              void* d_out, int out_size, void* d_ws, size_t ws_size,
                              hipStream_t stream) {
    const int*   inp = (const int*)  d_in[0];  // (128, 512)
    const float* emb = (const float*)d_in[1];  // (1, V, 300)
    const float* Wih = (const float*)d_in[2];  // (1024, 300)
    const float* Whh = (const float*)d_in[3];  // (1024, 256)
    const float* bih = (const float*)d_in[4];
    const float* bhh = (const float*)d_in[5];
    const float* W1  = (const float*)d_in[6];  // (128, 256)
    const float* b1  = (const float*)d_in[7];
    const float* W2  = (const float*)d_in[8];  // (1, 128)
    const float* b2  = (const float*)d_in[9];

    char* ws = (char*)d_ws;
    // offsets all multiples of 256B
    _Float16* Wcat    = (_Float16*)(ws);                       // 1,179,648 B
    float*    biasSum = (float*)   (ws + 1179648);             //     4,096 B
    _Float16* h0      = (_Float16*)(ws + 1183744);             //   262,144 B
    _Float16* h1      = (_Float16*)(ws + 1445888);             //   262,144 B
    float*    cbuf    = (float*)   (ws + 1708032);             //   524,288 B
    // total ~2.2 MB

    prep_wcat_kernel<<<(GATES * KTOT + 255) / 256, 256, 0, stream>>>(Wih, Whh, Wcat);
    prep_state_kernel<<<(SEQB * HID + 255) / 256, 256, 0, stream>>>(bih, bhh, biasSum,
                                                                    h0, h1, cbuf);

    _Float16* hb[2] = {h0, h1};
    for (int t = 0; t < 128; ++t) {
        lstm_step_kernel<<<64, 256, 0, stream>>>(inp + t * SEQB, emb, Wcat, biasSum,
                                                 hb[t & 1], hb[(t + 1) & 1], cbuf);
    }
    // after t=127, final h is in hb[0]
    mlp_kernel<<<2, 256, 0, stream>>>(h0, W1, b1, W2, b2, (float*)d_out);
}
